// Attention_87325275062587
// MI455X (gfx1250) — matrix operand order repaired and checked by reading, not compiled, not run on hardware
//
#include <hip/hip_runtime.h>
#include <hip/hip_bf16.h>

// ---------------------------------------------------------------------------
// GPT-2 attention block on gfx1250 (MI455X), f16 WMMA pipeline, f32 accum.
//   B=2, S=2048, D=1024, H=16, hd=64
// Register-blocked: GEMM waves own 32x32 output tiles (2x A/B reuse per
// WMMA); flash waves own 32-row query tiles (2x K/V fragment reuse).
// ---------------------------------------------------------------------------

#define Bv 2
#define Sv 2048
#define Dv 1024
#define Hv 16
#define HDv 64
#define Mv (Bv * Sv)      // 4096 rows of activations
#define NQKV (3 * Dv)     // 3072

typedef __attribute__((ext_vector_type(8)))  _Float16 v8h;
typedef __attribute__((ext_vector_type(16))) _Float16 v16h;
typedef __attribute__((ext_vector_type(8)))  float    v8f;

__device__ __forceinline__ v16h cat8(v8h a, v8h b) {
    return __builtin_shufflevector(a, b, 0,1,2,3,4,5,6,7,8,9,10,11,12,13,14,15);
}

// A-fragment (16x32 f16, M x K): lane = row (lane&15).
//   halves 0..7  : K = k0 + 8*(lane>>4) + 0..7
//   halves 8..15 : K = k0 + 16 + 8*(lane>>4) + 0..7
__device__ __forceinline__ v16h load_afrag(const _Float16* __restrict__ row,
                                           int k0, int hi) {
    const _Float16* p = row + k0 + hi * 8;
    return cat8(*(const v8h*)(p), *(const v8h*)(p + 16));
}

// B-fragment (32x16 f16, K x N) from a K-major (transposed) matrix Bt[N][K]:
//   lane = col (lane&15); halves 0..7 : K = k0 + 8*(lane>>4) + 0..7,
//   halves 8..15 : K = k0 + 16 + 8*(lane>>4) + 0..7
__device__ __forceinline__ v16h load_bfrag(const _Float16* __restrict__ row,
                                           int k0, int hi) {
    const _Float16* p = row + k0 + hi * 8;
    return cat8(*(const v8h*)(p), *(const v8h*)(p + 16));
}

__device__ __forceinline__ v8f wmma_f16(v16h a, v16h b, v8f c) {
    return __builtin_amdgcn_wmma_f32_16x16x32_f16(false, a, false, b,
                                                  (short)0, c, false, false);
}

// Row reductions across one 16-lane half of the wave (C-frag rows live in a
// single half: lanes 0-15 hold rows r, lanes 16-31 hold rows r+8).
__device__ __forceinline__ float hredmax(float v) {
#pragma unroll
    for (int m = 1; m < 16; m <<= 1) v = fmaxf(v, __shfl_xor(v, m, 32));
    return v;
}
__device__ __forceinline__ float hredsum(float v) {
#pragma unroll
    for (int m = 1; m < 16; m <<= 1) v += __shfl_xor(v, m, 32);
    return v;
}

// ---------------------------------------------------------------------------
// Prep kernels: f32 -> f16 convert, and weight transposes into [N][K] f16.
// ---------------------------------------------------------------------------
__global__ void k_cvt_f16(const float* __restrict__ in, _Float16* __restrict__ out, int n) {
    int i = blockIdx.x * blockDim.x + threadIdx.x;
    if (i < n) out[i] = (_Float16)in[i];
}

__global__ void k_transpose_wattn(const float* __restrict__ W, _Float16* __restrict__ Wt) {
    int i = blockIdx.x * blockDim.x + threadIdx.x;   // i = e*1024 + d
    if (i < NQKV * Dv) {
        int e = i >> 10, d = i & (Dv - 1);
        Wt[i] = (_Float16)W[d * NQKV + e];
    }
}

__global__ void k_transpose_wproj(const float* __restrict__ W, _Float16* __restrict__ Wt) {
    int i = blockIdx.x * blockDim.x + threadIdx.x;   // i = e*1024 + d
    if (i < Dv * Dv) {
        int e = i >> 10, d = i & (Dv - 1);
        Wt[i] = (_Float16)W[d * Dv + e];
    }
}

// ---------------------------------------------------------------------------
// QKV scatter epilogue: channel e of qkv row srow -> head-major Q/K, V^T.
// ---------------------------------------------------------------------------
__device__ __forceinline__ void qkv_scatter(float v, int e, int srow,
                                            _Float16* __restrict__ Q,
                                            _Float16* __restrict__ Kd,
                                            _Float16* __restrict__ Vt) {
    const int b = srow >> 11;
    const int s = srow & (Sv - 1);
    if (e < Dv) {                            // Q
        int h = e >> 6, d = e & 63;
        Q[((b * Hv + h) * Sv + s) * HDv + d] = (_Float16)v;
    } else if (e < 2 * Dv) {                 // K
        int e2 = e - Dv, h = e2 >> 6, d = e2 & 63;
        Kd[((b * Hv + h) * Sv + s) * HDv + d] = (_Float16)v;
    } else {                                 // V (store transposed)
        int e2 = e - 2 * Dv, h = e2 >> 6, d = e2 & 63;
        Vt[((b * Hv + h) * HDv + d) * Sv + s] = (_Float16)v;
    }
}

// ---------------------------------------------------------------------------
// QKV GEMM:  qkv[4096,3072] = x16[4096,1024] @ Wattn + b.
// Wave tile 32Mx32N (4 C frags, 2x A/B reuse); block = 8 waves = 64M x 128N.
// ---------------------------------------------------------------------------
__global__ void k_gemm_qkv(const _Float16* __restrict__ A,   // [4096][1024]
                           const _Float16* __restrict__ Bt,  // [3072][1024]
                           const float*    __restrict__ bias,
                           _Float16* __restrict__ Q,
                           _Float16* __restrict__ Kd,
                           _Float16* __restrict__ Vt) {
    const int lane  = threadIdx.x & 31;
    const int wave  = threadIdx.x >> 5;
    const int hi    = lane >> 4;
    const int lo    = lane & 15;
    const int m_sub = blockIdx.y * 64 + (wave & 1) * 32;
    const int n_sub = blockIdx.x * 128 + (wave >> 1) * 32;

    const _Float16* arow0 = A  + (m_sub + lo) * Dv;
    const _Float16* arow1 = arow0 + 16 * Dv;
    const _Float16* brow0 = Bt + (n_sub + lo) * Dv;
    const _Float16* brow1 = brow0 + 16 * Dv;

    v8f c00 = {}, c01 = {}, c10 = {}, c11 = {};
#pragma unroll 4
    for (int k0 = 0; k0 < Dv; k0 += 32) {
        __builtin_prefetch(arow0 + k0 + 512, 0, 3);
        __builtin_prefetch(brow0 + k0 + 512, 0, 3);
        v16h a0 = load_afrag(arow0, k0, hi);
        v16h a1 = load_afrag(arow1, k0, hi);
        v16h b0 = load_bfrag(brow0, k0, hi);
        v16h b1 = load_bfrag(brow1, k0, hi);
        c00 = wmma_f16(a0, b0, c00);
        c01 = wmma_f16(a0, b1, c01);
        c10 = wmma_f16(a1, b0, c10);
        c11 = wmma_f16(a1, b1, c11);
    }

    const int e0 = n_sub + lo;               // uniform region per wave
    const int e1 = e0 + 16;
    const float bb0 = bias[e0], bb1 = bias[e1];
#pragma unroll
    for (int r = 0; r < 8; r++) {
        const int s0 = m_sub + r + 8 * hi;
        const int s1 = s0 + 16;
        qkv_scatter(c00[r] + bb0, e0, s0, Q, Kd, Vt);
        qkv_scatter(c01[r] + bb1, e1, s0, Q, Kd, Vt);
        qkv_scatter(c10[r] + bb0, e0, s1, Q, Kd, Vt);
        qkv_scatter(c11[r] + bb1, e1, s1, Q, Kd, Vt);
    }
}

// ---------------------------------------------------------------------------
// Causal flash attention. One wave owns a 32-row query tile of one (b,h);
// kv processed 32 at a time: 8 score WMMAs reusing 4 K B-frags, online
// softmax per 16-row group, P relayout via per-wave LDS, 8 PV WMMAs reusing
// 4 V B-frags. Output f16 [B][S][D]. Block = 128 thr (4 waves).
// ---------------------------------------------------------------------------
__global__ void k_flash(const _Float16* __restrict__ Q,
                        const _Float16* __restrict__ Kd,
                        const _Float16* __restrict__ Vt,
                        _Float16* __restrict__ O) {
    __shared__ _Float16 pbuf[4][32 * 32];

    const int lane  = threadIdx.x & 31;
    const int wave  = threadIdx.x >> 5;
    const int hi    = lane >> 4;
    const int lo    = lane & 15;
    const int bh    = blockIdx.y;                 // b*16 + h
    const int qbase = (blockIdx.x * 4 + wave) * 32;
    const int q_hi  = qbase + 31;

    const _Float16* Qh = Q  + (size_t)bh * Sv * HDv;
    const _Float16* Kh = Kd + (size_t)bh * Sv * HDv;
    const _Float16* Vh = Vt + (size_t)bh * HDv * Sv;

    const _Float16* qrA = Qh + (qbase + lo) * HDv;       // rows qbase+0..15
    const _Float16* qrB = qrA + 16 * HDv;                // rows qbase+16..31
    const v16h aqA0 = load_afrag(qrA, 0, hi);
    const v16h aqA1 = load_afrag(qrA, 32, hi);
    const v16h aqB0 = load_afrag(qrB, 0, hi);
    const v16h aqB1 = load_afrag(qrB, 32, hi);

    v8f accA[4] = {}, accB[4] = {};
    float mA[8], lA[8], aA[8], mB[8], lB[8], aB[8];
#pragma unroll
    for (int r = 0; r < 8; r++) {
        mA[r] = -1e30f; lA[r] = 0.f;
        mB[r] = -1e30f; lB[r] = 0.f;
    }

    _Float16* pw = pbuf[wave];

    for (int kv0 = 0; kv0 <= q_hi; kv0 += 32) {
        // ---- scores: 2 row groups x 2 col tiles, reusing 4 K B-frags ----
        const _Float16* k0row = Kh + (kv0 + lo) * HDv;
        const _Float16* k1row = k0row + 16 * HDv;
        const v16h bk00 = load_bfrag(k0row, 0, hi);
        const v16h bk01 = load_bfrag(k0row, 32, hi);
        const v16h bk10 = load_bfrag(k1row, 0, hi);
        const v16h bk11 = load_bfrag(k1row, 32, hi);
        v8f sA0 = {}, sA1 = {}, sB0 = {}, sB1 = {};
        sA0 = wmma_f16(aqA0, bk00, sA0); sA0 = wmma_f16(aqA1, bk01, sA0);
        sA1 = wmma_f16(aqA0, bk10, sA1); sA1 = wmma_f16(aqA1, bk11, sA1);
        sB0 = wmma_f16(aqB0, bk00, sB0); sB0 = wmma_f16(aqB1, bk01, sB0);
        sB1 = wmma_f16(aqB0, bk10, sB1); sB1 = wmma_f16(aqB1, bk11, sB1);

        // ---- scale + causal mask + online softmax, row group A ----
#pragma unroll
        for (int r = 0; r < 8; r++) {
            const int row = qbase + r + 8 * hi;
            float v0 = sA0[r] * 0.125f;              // 1/sqrt(64)
            float v1 = sA1[r] * 0.125f;
            if (kv0 + lo > row)      v0 = -1e4f;     // reference masked value
            if (kv0 + 16 + lo > row) v1 = -1e4f;
            const float mt   = hredmax(fmaxf(v0, v1));
            const float newm = fmaxf(mA[r], mt);
            const float al   = __expf(mA[r] - newm);
            const float p0   = __expf(v0 - newm);
            const float p1   = __expf(v1 - newm);
            lA[r] = lA[r] * al + hredsum(p0 + p1);
            mA[r] = newm; aA[r] = al;
            pw[(r + 8 * hi) * 32 + lo]      = (_Float16)p0;
            pw[(r + 8 * hi) * 32 + 16 + lo] = (_Float16)p1;
        }
        // ---- row group B (rows qbase+16..31), staged in LDS rows 16..31 ----
#pragma unroll
        for (int r = 0; r < 8; r++) {
            const int row = qbase + 16 + r + 8 * hi;
            float v0 = sB0[r] * 0.125f;
            float v1 = sB1[r] * 0.125f;
            if (kv0 + lo > row)      v0 = -1e4f;
            if (kv0 + 16 + lo > row) v1 = -1e4f;
            const float mt   = hredmax(fmaxf(v0, v1));
            const float newm = fmaxf(mB[r], mt);
            const float al   = __expf(mB[r] - newm);
            const float p0   = __expf(v0 - newm);
            const float p1   = __expf(v1 - newm);
            lB[r] = lB[r] * al + hredsum(p0 + p1);
            mB[r] = newm; aB[r] = al;
            pw[(16 + r + 8 * hi) * 32 + lo]      = (_Float16)p0;
            pw[(16 + r + 8 * hi) * 32 + 16 + lo] = (_Float16)p1;
        }
        // cross-lane LDS round trip: make P stores visible before reload
        asm volatile("s_wait_dscnt 0" ::: "memory");
        const v16h pfA = load_afrag(pw + lo * 32, 0, hi);
        const v16h pfB = load_afrag(pw + (16 + lo) * 32, 0, hi);

        // ---- rescale accumulators, then PV WMMAs reusing 4 V B-frags ----
#pragma unroll
        for (int c = 0; c < 4; c++)
#pragma unroll
            for (int r = 0; r < 8; r++) {
                accA[c][r] *= aA[r];
                accB[c][r] *= aB[r];
            }
#pragma unroll
        for (int c = 0; c < 4; c++) {
            const v16h bv = load_bfrag(Vh + (c * 16 + lo) * Sv, kv0, hi);
            accA[c] = wmma_f16(pfA, bv, accA[c]);
            accB[c] = wmma_f16(pfB, bv, accB[c]);
        }
    }

    // ---- normalize and store merged-head output [B][S][D] f16 ----
    const int b = bh >> 4, h = bh & 15;
#pragma unroll
    for (int r = 0; r < 8; r++) {
        const int rowA = qbase + r + 8 * hi;
        const float iA = 1.0f / lA[r];
        const float iB = 1.0f / lB[r];
        _Float16* oA = O + ((size_t)(b * Sv + rowA)) * Dv + h * HDv + lo;
        _Float16* oB = oA + 16 * Dv;         // rowA + 16
#pragma unroll
        for (int c = 0; c < 4; c++) {
            oA[c * 16] = (_Float16)(accA[c][r] * iA);
            oB[c * 16] = (_Float16)(accB[c][r] * iB);
        }
    }
}

// ---------------------------------------------------------------------------
// Output projection: out[4096,1024] f32 = a16 @ Wproj_t + b_proj
// Wave tile 32Mx32N; block = 8 waves = 64M x 128N.
// ---------------------------------------------------------------------------
__global__ void k_gemm_proj(const _Float16* __restrict__ A,   // [4096][1024]
                            const _Float16* __restrict__ Bt,  // [1024][1024]
                            const float*    __restrict__ bias,
                            float* __restrict__ out) {
    const int lane  = threadIdx.x & 31;
    const int wave  = threadIdx.x >> 5;
    const int hi    = lane >> 4;
    const int lo    = lane & 15;
    const int m_sub = blockIdx.y * 64 + (wave & 1) * 32;
    const int n_sub = blockIdx.x * 128 + (wave >> 1) * 32;

    const _Float16* arow0 = A  + (m_sub + lo) * Dv;
    const _Float16* arow1 = arow0 + 16 * Dv;
    const _Float16* brow0 = Bt + (n_sub + lo) * Dv;
    const _Float16* brow1 = brow0 + 16 * Dv;

    v8f c00 = {}, c01 = {}, c10 = {}, c11 = {};
#pragma unroll 4
    for (int k0 = 0; k0 < Dv; k0 += 32) {
        __builtin_prefetch(arow0 + k0 + 512, 0, 3);
        __builtin_prefetch(brow0 + k0 + 512, 0, 3);
        v16h a0 = load_afrag(arow0, k0, hi);
        v16h a1 = load_afrag(arow1, k0, hi);
        v16h b0 = load_bfrag(brow0, k0, hi);
        v16h b1 = load_bfrag(brow1, k0, hi);
        c00 = wmma_f16(a0, b0, c00);
        c01 = wmma_f16(a0, b1, c01);
        c10 = wmma_f16(a1, b0, c10);
        c11 = wmma_f16(a1, b1, c11);
    }

    const float bb0 = bias[n_sub + lo];
    const float bb1 = bias[n_sub + 16 + lo];
#pragma unroll
    for (int r = 0; r < 8; r++) {
        const int s0 = m_sub + r + 8 * hi;
        const int s1 = s0 + 16;
        out[(size_t)s0 * Dv + n_sub + lo]      = c00[r] + bb0;
        out[(size_t)s0 * Dv + n_sub + 16 + lo] = c01[r] + bb1;
        out[(size_t)s1 * Dv + n_sub + lo]      = c10[r] + bb0;
        out[(size_t)s1 * Dv + n_sub + 16 + lo] = c11[r] + bb1;
    }
}

// ---------------------------------------------------------------------------
// Host launcher
// ---------------------------------------------------------------------------
extern "C" void kernel_launch(void* const* d_in, const int* in_sizes, int n_in,
                              void* d_out, int out_size, void* d_ws, size_t ws_size,
                              hipStream_t stream) {
    const float* x      = (const float*)d_in[0];   // [2,2048,1024]
    const float* W_attn = (const float*)d_in[1];   // [1024,3072]
    const float* b_attn = (const float*)d_in[2];   // [3072]
    const float* W_proj = (const float*)d_in[3];   // [1024,1024]
    const float* b_proj = (const float*)d_in[4];   // [1024]
    float* out = (float*)d_out;                    // [2,2048,1024]

    char* ws = (char*)d_ws;
    size_t off = 0;
    _Float16* x16  = (_Float16*)(ws + off); off += (size_t)Mv * Dv * 2;      // 8 MiB
    _Float16* WtA  = (_Float16*)(ws + off); off += (size_t)NQKV * Dv * 2;    // 6 MiB
    _Float16* WtP  = (_Float16*)(ws + off); off += (size_t)Dv * Dv * 2;      // 2 MiB
    _Float16* Qw   = (_Float16*)(ws + off); off += (size_t)Mv * Dv * 2;      // 8 MiB
    _Float16* Kw   = (_Float16*)(ws + off); off += (size_t)Mv * Dv * 2;      // 8 MiB
    _Float16* Vtw  = (_Float16*)(ws + off); off += (size_t)Mv * Dv * 2;      // 8 MiB
    _Float16* A16  = (_Float16*)(ws + off); off += (size_t)Mv * Dv * 2;      // 8 MiB
    (void)ws_size; (void)in_sizes; (void)n_in; (void)out_size;

    // 1) precision conversion + weight transposes
    {
        int n = Mv * Dv;
        k_cvt_f16<<<(n + 255) / 256, 256, 0, stream>>>(x, x16, n);
        int na = NQKV * Dv;
        k_transpose_wattn<<<(na + 255) / 256, 256, 0, stream>>>(W_attn, WtA);
        int np = Dv * Dv;
        k_transpose_wproj<<<(np + 255) / 256, 256, 0, stream>>>(W_proj, WtP);
    }

    // 2) fused QKV projection (WMMA) with head-split / V-transpose epilogue
    k_gemm_qkv<<<dim3(NQKV / 128, Mv / 64), 256, 0, stream>>>(
        x16, WtA, b_attn, Qw, Kw, Vtw);

    // 3) causal flash attention (WMMA scores + WMMA PV)
    k_flash<<<dim3(Sv / 128, Bv * Hv), 128, 0, stream>>>(Qw, Kw, Vtw, A16);

    // 4) output projection (WMMA), f32 result
    k_gemm_proj<<<dim3(Dv / 128, Mv / 64), 256, 0, stream>>>(
        A16, WtP, b_proj, out);
}
